// CustomLoraLinear_58136677319268
// MI455X (gfx1250) — matrix operand order repaired and checked by reading, not compiled, not run on hardware
//
#include <hip/hip_runtime.h>
#include <hip/hip_bf16.h>

// ---------------------------------------------------------------------------
// Problem constants
// ---------------------------------------------------------------------------
#define IN    4096
#define OUT   4096
#define RANK  16
#define GS    128
#define NG    32          // IN / GS
#define PF    8           // 32 / 4 bits
#define MTOT  8192        // 4 * 2048 rows of x
#define T     1.5f        // ternary threshold

// GEMM tiling
#define BM    128
#define BN    128
#define BK    32
#define LDSK  40          // padded LDS row length (bf16 elems); 80B rows, 16B aligned
#define KTILES (IN / BK)

// K1 chunking
#define CHUNK 256

typedef __attribute__((ext_vector_type(16))) __bf16       v16bf;
typedef __attribute__((ext_vector_type(8)))  float        v8f;
typedef __attribute__((ext_vector_type(4)))  float        f32x4;
typedef __attribute__((ext_vector_type(4)))  unsigned int u32x4;

// ---------------------------------------------------------------------------
// fp32 pair -> packed bf16.  Prefer V_CVT_PK_BF16_F32 if clang exposes it
// (probe said it does not on this toolchain); otherwise round-half-up via
// one v_add_nc_u32 per element + one v_perm_b32 per pair (3 VALU / pair).
// ---------------------------------------------------------------------------
#if __has_builtin(__builtin_amdgcn_cvt_pk_bf16_f32)
typedef __attribute__((ext_vector_type(2))) __bf16 v2bf;
__device__ __forceinline__ unsigned int pk2(float a, float b) {
    v2bf v = __builtin_amdgcn_cvt_pk_bf16_f32(a, b);
    return __builtin_bit_cast(unsigned int, v);
}
#else
__device__ __forceinline__ unsigned int pk2(float a, float b) {
    unsigned int ua = __float_as_uint(a) + 0x8000u;   // round half up
    unsigned int ub = __float_as_uint(b) + 0x8000u;
    // result bytes: [1:0] = ua[3:2], [3:2] = ub[3:2]  (v_perm_b32)
    return __builtin_amdgcn_perm(ub, ua, 0x07060302u);
}
#endif

// 8 floats -> 8 bf16 packed into 16B
__device__ __forceinline__ u32x4 pack8(f32x4 a, f32x4 b) {
    u32x4 r;
    r[0] = pk2(a[0], a[1]);
    r[1] = pk2(a[2], a[3]);
    r[2] = pk2(b[0], b[1]);
    r[3] = pk2(b[2], b[3]);
    return r;
}

struct Bf16FragBits { u32x4 lo, hi; };
__device__ __forceinline__ v16bf ld_frag(const unsigned short* p0,
                                         const unsigned short* p1) {
    Bf16FragBits b;
    b.lo = *(const u32x4*)p0;
    b.hi = *(const u32x4*)p1;
    return __builtin_bit_cast(v16bf, b);
}

// ---------------------------------------------------------------------------
// CDNA5 async global->LDS copy (ASYNCcnt-tracked, no VGPR staging).
// lds_addr: wave-relative LDS byte offset (low 32 bits of generic pointer).
// ---------------------------------------------------------------------------
__device__ __forceinline__ void async_b128(unsigned lds_addr, const void* gaddr) {
    asm volatile("global_load_async_to_lds_b128 %0, %1, off"
                 :: "v"(lds_addr), "v"(gaddr)
                 : "memory");
}
__device__ __forceinline__ void wait_async0() {
#if __has_builtin(__builtin_amdgcn_s_wait_asynccnt)
    __builtin_amdgcn_s_wait_asynccnt(0);
#else
    asm volatile("s_wait_asynccnt 0x0" ::: "memory");
#endif
}

// ---------------------------------------------------------------------------
// K1: per-column partial group sums of AB2 = AB - marker*T.
// grid = (OUT/256, IN/CHUNK), block = 256.  Each thread owns one output
// column o and a CHUNK of input rows; accumulates into LDS[g][tid] columns
// (conflict-free), then 32 global float atomics.
// ---------------------------------------------------------------------------
__global__ __launch_bounds__(256) void k_gsum(
    const float* __restrict__ lora_A,   // [RANK, IN]
    const float* __restrict__ lora_B,   // [OUT, RANK]
    const int*   __restrict__ qweight,  // [IN/PF, OUT]
    const int*   __restrict__ g_idx,    // [IN]
    float*       __restrict__ gsum)     // [NG, OUT]
{
    __shared__ float gs[NG][256];
    __shared__ int   gix[CHUNK];

    const int tid = threadIdx.x;
    const int o   = blockIdx.x * 256 + tid;
    const int i0  = blockIdx.y * CHUNK;

    gix[tid] = g_idx[i0 + tid];
#pragma unroll
    for (int g = 0; g < NG; ++g) gs[g][tid] = 0.f;
    __syncthreads();

    float bo[RANK];
#pragma unroll
    for (int r = 0; r < RANK; ++r) bo[r] = lora_B[(size_t)o * RANK + r];

    for (int ii = 0; ii < CHUNK; ii += 8) {
        const int i = i0 + ii;
        unsigned int qw = (unsigned int)qweight[(size_t)(i >> 3) * OUT + o];
#pragma unroll
        for (int j = 0; j < 8; ++j) {
            float ab = 0.f;
#pragma unroll
            for (int r = 0; r < RANK; ++r)
                ab += lora_A[(size_t)r * IN + i + j] * bo[r];
            const int w = (int)((qw >> (4 * j)) & 15u);
            const float mk = (ab >  T && w != 15) ?  1.f :
                             (ab < -T && w !=  0) ? -1.f : 0.f;
            gs[gix[ii + j]][tid] += ab - mk * T;
        }
    }
    __syncthreads();
#pragma unroll
    for (int g = 0; g < NG; ++g)
        atomicAdd(&gsum[(size_t)g * OUT + o], gs[g][tid]);
}

// ---------------------------------------------------------------------------
// K2: c1[g][o] = scales[g][o] * (zero_plus[g][o] - zeros[g][o])
//   zero_plus = gsum / (GS*T)
// ---------------------------------------------------------------------------
__global__ __launch_bounds__(256) void k_c1(
    const int*   __restrict__ qzeros,   // [NG, OUT/PF]
    const float* __restrict__ scales,   // [NG, OUT]
    const float* __restrict__ gsum,     // [NG, OUT]
    float*       __restrict__ c1)       // [NG, OUT]
{
    const int idx = blockIdx.x * 256 + threadIdx.x;   // = g*OUT + o
    const int g = idx >> 12;
    const int o = idx & (OUT - 1);
    unsigned int qz = (unsigned int)qzeros[(size_t)g * (OUT / PF) + (o >> 3)];
    const float z = (float)((qz >> (4 * (o & 7))) & 15u);
    c1[idx] = scales[idx] * (gsum[idx] * (1.0f / (GS * T)) - z);
}

// ---------------------------------------------------------------------------
// K3: build transposed bf16 weight  W[o][i] = scales*(w + marker) + c1.
// One thread per (o, 8 consecutive i): one qweight dword -> one 16B store.
// ---------------------------------------------------------------------------
__global__ __launch_bounds__(256) void k_build(
    const float* __restrict__ lora_A,
    const float* __restrict__ lora_B,
    const int*   __restrict__ qweight,
    const int*   __restrict__ g_idx,
    const float* __restrict__ scales,
    const float* __restrict__ c1,
    unsigned short* __restrict__ Wt)    // [OUT, IN] bf16
{
    const int gid = blockIdx.x * 256 + threadIdx.x;
    const int o   = gid >> 9;            // IN/PF = 512 i-chunks per column
    const int i8  = gid & 511;
    const int i0  = i8 * 8;

    float bo[RANK];
#pragma unroll
    for (int r = 0; r < RANK; ++r) bo[r] = lora_B[(size_t)o * RANK + r];

    unsigned int qw = (unsigned int)qweight[(size_t)i8 * OUT + o];
    float vals[8];
#pragma unroll
    for (int j = 0; j < 8; ++j) {
        const int i = i0 + j;
        float ab = 0.f;
#pragma unroll
        for (int r = 0; r < RANK; ++r)
            ab += lora_A[(size_t)r * IN + i] * bo[r];
        const int w = (int)((qw >> (4 * j)) & 15u);
        const float mk = (ab >  T && w != 15) ?  1.f :
                         (ab < -T && w !=  0) ? -1.f : 0.f;
        const int g = g_idx[i];
        vals[j] = scales[(size_t)g * OUT + o] * ((float)w + mk)
                + c1[(size_t)g * OUT + o];
    }
    u32x4 pk;
    pk[0] = pk2(vals[0], vals[1]);
    pk[1] = pk2(vals[2], vals[3]);
    pk[2] = pk2(vals[4], vals[5]);
    pk[3] = pk2(vals[6], vals[7]);
    *(u32x4*)(Wt + (size_t)o * IN + i0) = pk;
}

// ---------------------------------------------------------------------------
// K4: out[m][n] = sum_k x[m][k] * W[n][k]   (NT GEMM, bf16 WMMA, fp32 acc)
// 128x128 block tile, 8 wave32 per block (4x2 wave grid, each wave 32x64),
// BK=32, double-buffered LDS, v_wmma_f32_16x16x32_bf16.
// A (x, fp32) is staged via VGPRs + perm-pack bf16; B (Wt, bf16) is streamed
// straight into LDS with global_load_async_to_lds_b128 (ASYNCcnt).
// ---------------------------------------------------------------------------
__global__ __launch_bounds__(256) void k_gemm(
    const float*          __restrict__ x,    // [MTOT, IN] fp32
    const unsigned short* __restrict__ Wt,   // [OUT,  IN] bf16
    float*                __restrict__ out)  // [MTOT, OUT] fp32
{
    __shared__ unsigned short Al[2][BM * LDSK];
    __shared__ unsigned short Bl[2][BN * LDSK];

    const int tid = threadIdx.x;
    const int m0  = blockIdx.y * BM;
    const int n0  = blockIdx.x * BN;

    // Global->LDS staging: each thread owns 16 fp32 of x (one half-row)
    // and one async b128 (8 bf16) of W per K-step.
    const int row = tid >> 1;
    const int hf  = tid & 1;
    const float*          ax = x  + (size_t)(m0 + row) * IN + hf * 16;
    const unsigned short* bw = Wt + (size_t)(n0 + row) * IN + hf * 16;
    const int lrow = row * LDSK + hf * 16;
    const unsigned bl_lds[2] = {
        (unsigned)(unsigned long long)&Bl[0][lrow],
        (unsigned)(unsigned long long)&Bl[1][lrow]
    };

    // Wave/lane mapping for WMMA fragments
    const int lane = tid & 31;
    const int wid  = tid >> 5;
    const int wm   = (wid & 3) * 32;    // wave's M offset (2 tiles)
    const int wn   = (wid >> 2) * 64;   // wave's N offset (4 tiles)
    const int fr   = lane & 15;         // fragment row/col within tile
    const int h    = lane >> 4;         // lane half selects K sub-ranges

    v8f acc[2][4] = {};

    f32x4 fa0, fa1, fa2, fa3;

    // prologue: stage K-tile 0 (A via VGPRs, B via async)
    {
        const f32x4* p = (const f32x4*)ax;
        fa0 = p[0]; fa1 = p[1]; fa2 = p[2]; fa3 = p[3];
        async_b128(bl_lds[0], bw);
        *(u32x4*)&Al[0][lrow]     = pack8(fa0, fa1);
        *(u32x4*)&Al[0][lrow + 8] = pack8(fa2, fa3);
    }

    for (int kt = 0; kt < KTILES; ++kt) {
        wait_async0();                        // our async writes to this buf done
        __syncthreads();                      // everyone's stores/asyncs visible
        const int  buf  = kt & 1;
        const bool more = (kt + 1) < KTILES;

        if (more) {                           // overlap next-tile loads w/ compute
            async_b128(bl_lds[buf ^ 1], bw + (kt + 1) * BK);
            const f32x4* p = (const f32x4*)(ax + (kt + 1) * BK);
            fa0 = p[0]; fa1 = p[1]; fa2 = p[2]; fa3 = p[3];
        }

        v16bf afr[2], bfr[4];
#pragma unroll
        for (int tm = 0; tm < 2; ++tm) {
            // 16-bit A 16x32: lane<16 holds K {0..7,16..23}, lane>=16 {8..15,24..31}
            const unsigned short* p = &Al[buf][(wm + tm * 16 + fr) * LDSK];
            afr[tm] = ld_frag(p + h * 8, p + 16 + h * 8);
        }
#pragma unroll
        for (int tn = 0; tn < 4; ++tn) {
            // B 32x16: lane<16 holds K {0..7,16..23}, lane>=16 {8..15,24..31} (matches A)
            const unsigned short* p = &Bl[buf][(wn + tn * 16 + fr) * LDSK];
            bfr[tn] = ld_frag(p + h * 8, p + 16 + h * 8);
        }
#pragma unroll
        for (int tm = 0; tm < 2; ++tm)
#pragma unroll
            for (int tn = 0; tn < 4; ++tn)
                acc[tm][tn] = __builtin_amdgcn_wmma_f32_16x16x32_bf16(
                    false, afr[tm], false, bfr[tn],
                    (short)0, acc[tm][tn], false, false);

        if (more) {                           // fill the other A buffer
            const int nb = buf ^ 1;
            *(u32x4*)&Al[nb][lrow]     = pack8(fa0, fa1);
            *(u32x4*)&Al[nb][lrow + 8] = pack8(fa2, fa3);
        }
    }

    // epilogue: C/D layout — lanes 0-15: M=v, lanes 16-31: M=v+8; N = lane&15
#pragma unroll
    for (int tm = 0; tm < 2; ++tm)
#pragma unroll
        for (int tn = 0; tn < 4; ++tn) {
            float* o = out + (size_t)(m0 + wm + tm * 16 + h * 8) * OUT
                           + (n0 + wn + tn * 16 + fr);
#pragma unroll
            for (int v = 0; v < 8; ++v)
                o[(size_t)v * OUT] = acc[tm][tn][v];
        }
}

// ---------------------------------------------------------------------------
// Launch
// ---------------------------------------------------------------------------
extern "C" void kernel_launch(void* const* d_in, const int* in_sizes, int n_in,
                              void* d_out, int out_size, void* d_ws, size_t ws_size,
                              hipStream_t stream) {
    const float* x       = (const float*)d_in[0];
    const float* lora_A  = (const float*)d_in[1];
    const float* lora_B  = (const float*)d_in[2];
    const int*   qweight = (const int*)  d_in[3];
    const int*   qzeros  = (const int*)  d_in[4];
    const float* scales  = (const float*)d_in[5];
    const int*   g_idx   = (const int*)  d_in[6];
    float*       out     = (float*)d_out;

    char* ws = (char*)d_ws;
    float*          gsum = (float*)ws;                               // 512 KB
    float*          c1   = (float*)(ws + (size_t)NG * OUT * 4);      // 512 KB
    unsigned short* Wt   = (unsigned short*)(ws + 2 * (size_t)NG * OUT * 4); // 32 MB

    hipMemsetAsync(gsum, 0, (size_t)NG * OUT * sizeof(float), stream);

    k_gsum<<<dim3(OUT / 256, IN / CHUNK), 256, 0, stream>>>(
        lora_A, lora_B, qweight, g_idx, gsum);

    k_c1<<<(NG * OUT) / 256, 256, 0, stream>>>(qzeros, scales, gsum, c1);

    k_build<<<(OUT * (IN / PF)) / 256, 256, 0, stream>>>(
        lora_A, lora_B, qweight, g_idx, scales, c1, Wt);

    k_gemm<<<dim3(OUT / BN, MTOT / BM), 256, 0, stream>>>(x, Wt, out);
}
